// TransformerEncoderLayer_51685636440169
// MI455X (gfx1250) — matrix operand order repaired and checked by reading, not compiled, not run on hardware
//
#include <hip/hip_runtime.h>

// ---------------- problem constants ----------------
constexpr int B_  = 4;
constexpr int S_  = 4096;
constexpr int DM  = 512;
constexpr int DK  = 32;
constexpr int M_  = B_ * S_;       // 16384 rows
constexpr float NINF = -1.0e24f;
constexpr float EPS  = 1e-5f;

typedef __attribute__((ext_vector_type(16))) __bf16 v16bf;
typedef __attribute__((ext_vector_type(8)))  __bf16 v8bf;
typedef __attribute__((ext_vector_type(4)))  __bf16 v4bf;
typedef __attribute__((ext_vector_type(8)))  float  v8f;

#define DEVI static __device__ __forceinline__

// ---------------- WMMA helpers ----------------
// A fragment (16x32 bf16, row-major, one row per lane&15):
//   elements 0..15 -> K = 16*hi + 0..15 (contiguous)
DEVI v16bf afrag_load(const __bf16* row, int hi) {
    union { v16bf v; v8bf h[2]; } u;
    u.h[0] = *reinterpret_cast<const v8bf*>(row + 16 * hi);
    u.h[1] = *reinterpret_cast<const v8bf*>(row + 16 * hi + 8);
    return u.v;
}
// B fragment (32x16 bf16, column per lane&15, K-major per column):
//   elements 0..15 -> K = 16*hi + 0..15 (contiguous)
DEVI v16bf bfrag_load(const __bf16* colK0, int hi) {
    union { v16bf v; v8bf h[2]; } u;
    const __bf16* p = colK0 + 16 * hi;
    u.h[0] = *reinterpret_cast<const v8bf*>(p);
    u.h[1] = *reinterpret_cast<const v8bf*>(p + 8);
    return u.v;
}
DEVI v8f wmma_bf16(v16bf a, v16bf b, v8f c) {
    return __builtin_amdgcn_wmma_f32_16x16x32_bf16(false, a, false, b,
                                                   (short)0, c, false, false);
}

// ---------------- K0: f32 -> bf16 convert ----------------
__global__ void k_convert(const float* __restrict__ in, __bf16* __restrict__ out, size_t n) {
    size_t i = (size_t)blockIdx.x * blockDim.x + threadIdx.x;
    size_t stride = (size_t)gridDim.x * blockDim.x;
    for (; i < n; i += stride) out[i] = (__bf16)in[i];
}

// ---------------- K1: projection GEMM  out[M,N] = bf16( scale * (A[M,512] @ W[512,N]) ) ----
// MODE 0: plain; MODE 1: scale (for Q); MODE 2: key-padding NINF rows (for K)
template <int MODE>
__global__ __launch_bounds__(256) void k_proj_gemm(const __bf16* __restrict__ A,
                                                   const float* __restrict__ W,
                                                   __bf16* __restrict__ out,
                                                   int N,
                                                   const int* __restrict__ kpm,
                                                   float scale) {
    const int wave = threadIdx.x >> 5;
    const int lane = threadIdx.x & 31;
    const int ln = lane & 15, hi = lane >> 4;
    const int tile = blockIdx.x * 8 + wave;
    const int ntiles = N >> 4;
    const int m0 = (tile / ntiles) << 4;
    const int n0 = (tile % ntiles) << 4;

    v8f acc = {};
    const __bf16* arow = A + (size_t)(m0 + ln) * DM;
    for (int k0 = 0; k0 < DM; k0 += 32) {
        v16bf a = afrag_load(arow + k0, hi);
        v16bf b;
        #pragma unroll
        for (int i = 0; i < 16; ++i) {
            int kk = k0 + 16 * hi + i;
            b[i] = (__bf16)W[(size_t)kk * N + n0 + ln];
        }
        acc = wmma_bf16(a, b, acc);
    }
    #pragma unroll
    for (int r = 0; r < 8; ++r) {
        int row = m0 + r + 8 * hi;
        float v = acc[r];
        if (MODE == 1) v *= scale;
        if (MODE == 2) v = kpm[row] ? NINF : v;
        out[(size_t)row * N + n0 + ln] = (__bf16)v;
    }
}

// ---------------- K2: flash attention ----------------
// grid: B*(S/32) blocks, 256 threads. wave: qt = wave>>2 (16-row q tile), cg = wave&3 (128 cols)
__global__ __launch_bounds__(256) void k_flash(const __bf16* __restrict__ Qb,
                                               const __bf16* __restrict__ Kb,
                                               const __bf16* __restrict__ Vb,
                                               const int* __restrict__ amask,
                                               float* __restrict__ hbuf) {
    __shared__ __bf16 Kt[32 * 32];        // [key][d]  pitch 32 (linear copy target)
    __shared__ __bf16 Vt[512 * 40];       // [col][key] pitch 40 (80B, 8/16B aligned)
    __shared__ __bf16 Pw[8][16 * 32];     // per-wave P tile [m][key], pitch 32

    const int tid = threadIdx.x;
    const int wave = tid >> 5, lane = tid & 31;
    const int ln = lane & 15, hi = lane >> 4;
    const int b  = blockIdx.x / (S_ / 32);
    const int q0 = (blockIdx.x % (S_ / 32)) * 32;
    const int qt = wave >> 2, cg = wave & 3;
    const int qbase = q0 + qt * 16;

    // Q fragment (1/sqrt(dk) already folded into Qb)
    v16bf aq = afrag_load(Qb + (size_t)(b * S_ + qbase + ln) * DK, hi);

    v8f hacc[8];
    #pragma unroll
    for (int t = 0; t < 8; ++t) hacc[t] = (v8f){};
    float mstat[8], lstat[8];
    #pragma unroll
    for (int r = 0; r < 8; ++r) { mstat[r] = -3.0e38f; lstat[r] = 0.0f; }

    // per-thread constants for async K staging (2048B tile = 8B per thread)
    const unsigned ktLds = (unsigned)(uintptr_t)(&Kt[0]) + tid * 8;

    for (int kb = 0; kb < S_; kb += 32) {
        __syncthreads();
        // --- stage K tile via async global->LDS DMA (CDNA5 path, ASYNCcnt) ---
        {
            const __bf16* g = Kb + (size_t)(b * S_ + kb) * DK + tid * 4;
            asm volatile("global_load_async_to_lds_b64 %0, %1, off"
                         :: "v"(ktLds), "v"(g) : "memory");
        }
        // --- stage V tile transposed: Vt[col][key], 4 keys packed per b64 store ---
        for (int ch = tid; ch < 512; ch += 256) {
            int k4 = (ch & 7) << 2;         // key base (0..28)
            int c8 = (ch >> 3) << 3;        // col base (0..504)
            const __bf16* vrow = Vb + (size_t)(b * S_ + kb + k4) * DM + c8;
            v8bf r0 = *reinterpret_cast<const v8bf*>(vrow);
            v8bf r1 = *reinterpret_cast<const v8bf*>(vrow + DM);
            v8bf r2 = *reinterpret_cast<const v8bf*>(vrow + 2 * DM);
            v8bf r3 = *reinterpret_cast<const v8bf*>(vrow + 3 * DM);
            #pragma unroll
            for (int j = 0; j < 8; ++j) {
                v4bf p; p[0] = r0[j]; p[1] = r1[j]; p[2] = r2[j]; p[3] = r3[j];
                *reinterpret_cast<v4bf*>(&Vt[(c8 + j) * 40 + k4]) = p;
            }
        }
        asm volatile("s_wait_asynccnt 0x0" ::: "memory");
        __syncthreads();

        // QK^T: two 16-key tiles
        v8f s0 = {}, s1 = {};
        s0 = wmma_bf16(aq, bfrag_load(&Kt[(ln) * 32], hi), s0);
        s1 = wmma_bf16(aq, bfrag_load(&Kt[(16 + ln) * 32], hi), s1);

        // mask + online softmax per row r (row = r + 8*hi, cols across 16 lanes)
        #pragma unroll
        for (int r = 0; r < 8; ++r) {
            int qr = qbase + r + 8 * hi;
            const int* mrow = amask + ((size_t)(b * S_ + qr)) * S_ + kb;
            float x0 = mrow[ln]      ? NINF : s0[r];
            float x1 = mrow[16 + ln] ? NINF : s1[r];
            float mx = fmaxf(x0, x1);
            #pragma unroll
            for (int m = 1; m < 16; m <<= 1) mx = fmaxf(mx, __shfl_xor(mx, m, 32));
            float mnew = fmaxf(mstat[r], mx);
            float p0 = __expf(x0 - mnew), p1 = __expf(x1 - mnew);
            float rs = p0 + p1;
            #pragma unroll
            for (int m = 1; m < 16; m <<= 1) rs += __shfl_xor(rs, m, 32);
            float alpha = __expf(mstat[r] - mnew);
            lstat[r] = lstat[r] * alpha + rs;
            mstat[r] = mnew;
            #pragma unroll
            for (int t = 0; t < 8; ++t) hacc[t][r] *= alpha;
            int mrow16 = (r + 8 * hi) * 32;
            Pw[wave][mrow16 + ln]      = (__bf16)p0;
            Pw[wave][mrow16 + 16 + ln] = (__bf16)p1;
        }
        __syncthreads();

        // P @ V for this wave's 128 columns
        v16bf ap = afrag_load(&Pw[wave][ln * 32], hi);
        #pragma unroll
        for (int t = 0; t < 8; ++t) {
            v16bf bv = bfrag_load(&Vt[(cg * 128 + t * 16 + ln) * 40], hi);
            hacc[t] = wmma_bf16(ap, bv, hacc[t]);
        }
    }

    // normalize and write h
    #pragma unroll
    for (int t = 0; t < 8; ++t) {
        #pragma unroll
        for (int r = 0; r < 8; ++r) {
            int qr = qbase + r + 8 * hi;
            int col = cg * 128 + t * 16 + ln;
            hbuf[((size_t)(b * S_ + qr)) * DM + col] = hacc[t][r] / lstat[r];
        }
    }
}

// ---------------- K3/K5: residual add + LayerNorm (wave per row) ----------------
__global__ __launch_bounds__(256) void k_add_ln(const float* __restrict__ X,
                                                const float* __restrict__ H,
                                                const float* __restrict__ g,
                                                const float* __restrict__ bta,
                                                float* __restrict__ zout,
                                                __bf16* __restrict__ zbout) {
    const int wave = threadIdx.x >> 5, lane = threadIdx.x & 31;
    const int row = blockIdx.x * 8 + wave;
    const float* xr = X + (size_t)row * DM;
    const float* hr = H ? H + (size_t)row * DM : nullptr;

    float vals[16];
    float s1 = 0.f, s2 = 0.f;
    #pragma unroll
    for (int j = 0; j < 16; ++j) {
        int c = lane + 32 * j;
        float v = xr[c] + (hr ? hr[c] : 0.0f);
        vals[j] = v; s1 += v; s2 += v * v;
    }
    #pragma unroll
    for (int m = 1; m < 32; m <<= 1) { s1 += __shfl_xor(s1, m, 32); s2 += __shfl_xor(s2, m, 32); }
    float mu = s1 * (1.0f / DM);
    float var = s2 * (1.0f / DM) - mu * mu;
    float rs = rsqrtf(var + EPS);
    #pragma unroll
    for (int j = 0; j < 16; ++j) {
        int c = lane + 32 * j;
        float z = (vals[j] - mu) * rs * g[c] + bta[c];
        zout[(size_t)row * DM + c] = z;
        if (zbout) zbout[(size_t)row * DM + c] = (__bf16)z;
    }
}

// ---------------- K4: FFN GEMM  t = z + leaky_relu(zb @ Wf + bf) ----------------
__global__ __launch_bounds__(256) void k_ffn_gemm(const __bf16* __restrict__ A,
                                                  const float* __restrict__ W,
                                                  const float* __restrict__ bias,
                                                  const float* __restrict__ z,
                                                  float* __restrict__ tout) {
    const int wave = threadIdx.x >> 5, lane = threadIdx.x & 31;
    const int ln = lane & 15, hi = lane >> 4;
    const int tile = blockIdx.x * 8 + wave;
    const int ntiles = DM >> 4;                // 32
    const int m0 = (tile / ntiles) << 4;
    const int n0 = (tile % ntiles) << 4;

    v8f acc = {};
    const __bf16* arow = A + (size_t)(m0 + ln) * DM;
    for (int k0 = 0; k0 < DM; k0 += 32) {
        v16bf a = afrag_load(arow + k0, hi);
        v16bf b;
        #pragma unroll
        for (int i = 0; i < 16; ++i) {
            int kk = k0 + 16 * hi + i;
            b[i] = (__bf16)W[(size_t)kk * DM + n0 + ln];
        }
        acc = wmma_bf16(a, b, acc);
    }
    #pragma unroll
    for (int r = 0; r < 8; ++r) {
        int row = m0 + r + 8 * hi;
        int col = n0 + ln;
        float v = acc[r] + bias[col];
        float y = v > 0.0f ? v : 0.01f * v;
        tout[(size_t)row * DM + col] = z[(size_t)row * DM + col] + y;
    }
}

// ---------------- host launcher ----------------
extern "C" void kernel_launch(void* const* d_in, const int* in_sizes, int n_in,
                              void* d_out, int out_size, void* d_ws, size_t ws_size,
                              hipStream_t stream) {
    (void)in_sizes; (void)n_in; (void)out_size; (void)ws_size;
    const float* x   = (const float*)d_in[0];
    const int*  amask= (const int*)  d_in[1];
    const int*  kpm  = (const int*)  d_in[2];
    const float* Wq  = (const float*)d_in[3];
    const float* Wk  = (const float*)d_in[4];
    const float* Wv  = (const float*)d_in[5];
    const float* Wf  = (const float*)d_in[6];
    const float* bfv = (const float*)d_in[7];
    const float* g1  = (const float*)d_in[8];
    const float* b1  = (const float*)d_in[9];
    const float* g2  = (const float*)d_in[10];
    const float* b2  = (const float*)d_in[11];
    float* out = (float*)d_out;

    char* ws = (char*)d_ws;
    const size_t SZ_XB = (size_t)M_ * DM * 2;   // 16.8 MB
    const size_t SZ_QK = (size_t)M_ * DK * 2;   // 1 MB
    const size_t SZ_F  = (size_t)M_ * DM * 4;   // 33.6 MB
    __bf16* xb = (__bf16*)ws;                              // later reused as zb16
    __bf16* Qb = (__bf16*)(ws + SZ_XB);
    __bf16* Kb = (__bf16*)(ws + SZ_XB + SZ_QK);
    __bf16* Vb = (__bf16*)(ws + SZ_XB + 2 * SZ_QK);
    float*  hb = (float*) (ws + SZ_XB + 2 * SZ_QK + SZ_XB); // later reused as t
    float*  zb_f32 = (float*)(ws + SZ_XB + 2 * SZ_QK + SZ_XB + SZ_F);
    __bf16* zb16 = xb;       // alias: xb dead after projections
    float*  tb   = hb;       // alias: h dead after first LN

    const float invdk = 0.17677669529663687f;  // 1/sqrt(32)

    k_convert<<<8192, 256, 0, stream>>>(x, xb, (size_t)M_ * DM);

    // projections: (M/16)*(N/16) tiles, 8 tiles per block
    k_proj_gemm<1><<<(M_ / 16) * (DK / 16) / 8, 256, 0, stream>>>(xb, Wq, Qb, DK, nullptr, invdk);
    k_proj_gemm<2><<<(M_ / 16) * (DK / 16) / 8, 256, 0, stream>>>(xb, Wk, Kb, DK, kpm, 1.0f);
    k_proj_gemm<0><<<(M_ / 16) * (DM / 16) / 8, 256, 0, stream>>>(xb, Wv, Vb, DM, nullptr, 1.0f);

    // flash attention
    k_flash<<<B_ * (S_ / 32), 256, 0, stream>>>(Qb, Kb, Vb, amask, hb);

    // z = LN(x + h)
    k_add_ln<<<M_ / 8, 256, 0, stream>>>(x, hb, g1, b1, zb_f32, zb16);

    // t = z + leaky(zb @ Wf + bf)
    k_ffn_gemm<<<(M_ / 16) * (DM / 16) / 8, 256, 0, stream>>>(zb16, Wf, bfv, zb_f32, tb);

    // out = LN(t)
    k_add_ln<<<M_ / 8, 256, 0, stream>>>(tb, nullptr, g2, b2, out, nullptr);
}
